// VimBlock_50242527428690
// MI455X (gfx1250) — matrix operand order repaired and checked by reading, not compiled, not run on hardware
//
#include <hip/hip_runtime.h>

// ---------------------------------------------------------------------------
// Bidirectional Mamba (Vim) block for MI455X (gfx1250, wave32, WMMA + TDM).
// Projections run on v_wmma_f32_16x16x32_bf16; A/B tiles staged to LDS by the
// Tensor Data Mover (tensor_load_to_lds), double-buffered on TENSORcnt.
// ---------------------------------------------------------------------------

typedef __attribute__((ext_vector_type(16))) __bf16 v16bf;
typedef __attribute__((ext_vector_type(8)))  float  v8f;
typedef __attribute__((ext_vector_type(4)))  unsigned int v4u;
typedef __attribute__((ext_vector_type(8)))  int v8i_t;
typedef __attribute__((ext_vector_type(4)))  int v4i_t;

#define BATCH   4
#define SEQLEN  2048
#define DMODEL  1024
#define DINNER  2048
#define DTRANK  64
#define DSTATE  16
#define MROWS   (BATCH * SEQLEN)          // 8192
#define XDBLN   (DTRANK + 2 * DSTATE)     // 96

// ---------------------------------------------------------------------------
// TDM: issue a 2D tensor_load_to_lds of a tile_d1 x tile_d0 (rows x elems)
// bf16 tile from global (row stride = stride_elems) into LDS at lds_byte_addr.
// D# layout per CDNA5 ISA 8.3/8.4: group0 = {ctrl, lds_addr, global_addr,
// type=2}, group1 = {data_size=2B, tensor dims, tile dims, dim0 stride}.
// Rows/elems beyond tensor_dim return zero (used for the N=96 GEMM).
// ---------------------------------------------------------------------------
__device__ __forceinline__ void tdm_load_2d(unsigned lds_byte_addr,
                                            const __bf16* gptr,
                                            unsigned tile_d0, unsigned tile_d1,
                                            unsigned stride_elems,
                                            unsigned tdim0, unsigned tdim1) {
  const unsigned long long ga = (unsigned long long)(uintptr_t)gptr;
  v4u g0;
  g0[0] = 1u;                                   // count=1 (valid), user mode
  g0[1] = lds_byte_addr;                        // lds_addr [63:32]
  g0[2] = (unsigned)ga;                         // global_addr low
  g0[3] = ((unsigned)(ga >> 32) & 0x01ffffffu)  // global_addr high (57-bit)
          | (2u << 30);                         // type = 2 ("image")
  v8i_t g1;
  g1[0] = 0x10000;                              // data_size=1 (2 bytes)
  g1[1] = (int)((tdim0 & 0xffffu) << 16);       // tensor_dim0[15:0] @ bits 63:48
  g1[2] = (int)(((tdim0 >> 16) & 0xffffu) | ((tdim1 & 0xffffu) << 16));
  g1[3] = (int)(((tdim1 >> 16) & 0xffffu) | ((tile_d0 & 0xffffu) << 16));
  g1[4] = (int)(tile_d1 & 0xffffu);             // tile_dim1; tile_dim2=0
  g1[5] = (int)stride_elems;                    // tensor_dim0_stride[31:0]
  g1[6] = 0;                                    // stride[47:32]=0, dim1_stride lo
  g1[7] = 0;                                    // dim1_stride hi
  const v4i_t z4 = {0, 0, 0, 0};
  const v8i_t z8 = {0, 0, 0, 0, 0, 0, 0, 0};
  __builtin_amdgcn_tensor_load_to_lds(g0, g1, z4, z4, z8, 0);
}

// ---------------------------------------------------------------------------
// LayerNorm -> bf16 (one block per row of 1024)
// ---------------------------------------------------------------------------
__global__ __launch_bounds__(256)
void ln_bf16_kernel(const float* __restrict__ x, const float* __restrict__ w,
                    const float* __restrict__ b, __bf16* __restrict__ out) {
  const int row = blockIdx.x;
  const float* xr = x + (size_t)row * DMODEL;
  float v[4];
  float s = 0.f;
#pragma unroll
  for (int i = 0; i < 4; ++i) { v[i] = xr[threadIdx.x + 256 * i]; s += v[i]; }
  __shared__ float red[256];
  red[threadIdx.x] = s; __syncthreads();
  for (int off = 128; off > 0; off >>= 1) {
    if (threadIdx.x < off) red[threadIdx.x] += red[threadIdx.x + off];
    __syncthreads();
  }
  const float mu = red[0] * (1.f / DMODEL);
  __syncthreads();
  float s2 = 0.f;
#pragma unroll
  for (int i = 0; i < 4; ++i) { float d = v[i] - mu; s2 += d * d; }
  red[threadIdx.x] = s2; __syncthreads();
  for (int off = 128; off > 0; off >>= 1) {
    if (threadIdx.x < off) red[threadIdx.x] += red[threadIdx.x + off];
    __syncthreads();
  }
  const float rstd = rsqrtf(red[0] * (1.f / DMODEL) + 1e-5f);
#pragma unroll
  for (int i = 0; i < 4; ++i) {
    const int c = threadIdx.x + 256 * i;
    out[(size_t)row * DMODEL + c] = (__bf16)((v[i] - mu) * rstd * w[c] + b[c]);
  }
}

// ---------------------------------------------------------------------------
// f32 -> bf16 conversion
// ---------------------------------------------------------------------------
__global__ void cvt_bf16_kernel(const float* __restrict__ src,
                                __bf16* __restrict__ dst, int n) {
  const int i = blockIdx.x * blockDim.x + threadIdx.x;
  if (i < n) dst[i] = (__bf16)src[i];
}

// ---------------------------------------------------------------------------
// WMMA GEMM:  C[M,N] (+)= alpha * A[M,K] * B[N,K]^T     (bf16 in, f32 out)
// Block = 256 threads = 8 waves; block tile 128x128; k-step 32.
// A/B tiles (128x32 bf16, 8KB each) staged to LDS by TDM from wave 0,
// double-buffered with s_wait_tensorcnt; all waves read fragments via
// ds_load_b128 in the ISA 7.12.2 16-bit layouts.  sched_barriers force
// all 18 ds_loads to issue before the 8 back-to-back WMMAs (one dscnt wait).
// ---------------------------------------------------------------------------
union FragAB { v16bf v; int4 q[2]; };

__global__ __launch_bounds__(256)
void wmma_gemm_kernel(const __bf16* __restrict__ A, const __bf16* __restrict__ B,
                      float* __restrict__ C, int M, int N, int K,
                      int lda, int ldb, int ldc, float alpha, int accum) {
  __shared__ __bf16 sA[2][128 * 32];   // [row][k], row stride 32 elems (64B)
  __shared__ __bf16 sB[2][128 * 32];   // [col][k]

  const int lane = threadIdx.x & 31;
  const int wave = threadIdx.x >> 5;
  const int half = lane >> 4;
  const int l16  = lane & 15;
  const int nbase = blockIdx.x * 128;
  const int mbase = blockIdx.y * 128;
  const int nK = K >> 5;

  auto issue = [&](int ki, int buf) {
    const int kb = ki * 32;
    tdm_load_2d((unsigned)(uintptr_t)&sA[buf][0],
                A + (size_t)mbase * lda + kb, 32, 128, (unsigned)lda,
                (unsigned)(K - kb), (unsigned)(M - mbase));
    tdm_load_2d((unsigned)(uintptr_t)&sB[buf][0],
                B + (size_t)nbase * ldb + kb, 32, 128, (unsigned)ldb,
                (unsigned)(K - kb), (unsigned)(N - nbase));
  };

  v8f accv[8];
  const v8f vz = {0.f, 0.f, 0.f, 0.f, 0.f, 0.f, 0.f, 0.f};
#pragma unroll
  for (int j = 0; j < 8; ++j) accv[j] = vz;

  if (wave == 0) issue(0, 0);

  for (int ki = 0; ki < nK; ++ki) {
    const int cur = ki & 1;
    if (wave == 0) {
      if (ki + 1 < nK) {
        issue(ki + 1, cur ^ 1);                  // prefetch next tiles
        __builtin_amdgcn_s_wait_tensorcnt(2);    // current pair landed
      } else {
        __builtin_amdgcn_s_wait_tensorcnt(0);
      }
    }
    __syncthreads();                             // tiles visible to all waves

    const __bf16* As = &sA[cur][0];
    const __bf16* Bs = &sB[cur][0];

    // A fragment 16x32: half 0 -> K {0..7, 16..23}; half 1 -> {8..15, 24..31}
    FragAB a;
    const int arow = wave * 16 + l16;
    a.q[0] = *(const int4*)(As + arow * 32 + half * 8);
    a.q[1] = *(const int4*)(As + arow * 32 + 16 + half * 8);

    // Preload all 8 B fragments (32x16: lanes 0-15 K {0..7,16..23}, 16-31 K {8..15,24..31})
    FragAB bfr[8];
#pragma unroll
    for (int j = 0; j < 8; ++j) {
      const __bf16* bp = Bs + (j * 16 + l16) * 32 + half * 8;
      bfr[j].q[0] = *(const int4*)(bp);
      bfr[j].q[1] = *(const int4*)(bp + 16);
    }

    // Fence: all ds_loads above must issue before any WMMA below.
    __builtin_amdgcn_sched_barrier(0);

#pragma unroll
    for (int j = 0; j < 8; ++j)
      accv[j] = __builtin_amdgcn_wmma_f32_16x16x32_bf16(
          false, a.v, false, bfr[j].v, (short)0, accv[j], false, false);

    __builtin_amdgcn_sched_barrier(0);

    __syncthreads();                             // protect buffer for reuse
  }

  // C layout: element i of v8f -> row = mwave + half*8 + i, col = tile + l16
  const int crow0 = mbase + wave * 16 + half * 8;
#pragma unroll
  for (int j = 0; j < 8; ++j) {
    const int ncol = nbase + j * 16 + l16;
    if (ncol >= N) continue;
#pragma unroll
    for (int i = 0; i < 8; ++i) {
      const int row = crow0 + i;
      if (row >= M) continue;
      const size_t off = (size_t)row * ldc + ncol;
      float v = alpha * accv[j][i];
      if (accum) v += C[off];
      C[off] = v;
    }
  }
}

// ---------------------------------------------------------------------------
// Depthwise causal conv (K=4) + bias + SiLU.  dir==0: causal (forward),
// dir==1: anti-causal with reversed taps (equivalent to flip-conv-flip).
// ---------------------------------------------------------------------------
__global__ void conv_silu_kernel(const float* __restrict__ xz,
                                 const float* __restrict__ Wc,
                                 const float* __restrict__ bc,
                                 float* __restrict__ xc,
                                 __bf16* __restrict__ xcbf, int dir) {
  const int idx = blockIdx.x * blockDim.x + threadIdx.x;
  if (idx >= MROWS * DINNER) return;
  const int ch = idx & (DINNER - 1);
  const int r  = idx >> 11;
  const int t  = r & (SEQLEN - 1);
  const int b  = r >> 11;
  float acc = bc[ch];
#pragma unroll
  for (int k = 0; k < 4; ++k) {
    int tt = dir ? (t + 3 - k) : (t - 3 + k);
    if (tt >= 0 && tt < SEQLEN)
      acc += Wc[ch * 4 + k] * xz[((size_t)((b << 11) + tt)) * (2 * DINNER) + ch];
  }
  const float s = acc / (1.f + __expf(-acc));   // SiLU
  xc[idx] = s;
  xcbf[idx] = (__bf16)s;
}

// ---------------------------------------------------------------------------
// dt = softplus(dt_pre + b_dt), in place
// ---------------------------------------------------------------------------
__global__ void softplus_kernel(float* __restrict__ dt,
                                const float* __restrict__ bdt) {
  const int idx = blockIdx.x * blockDim.x + threadIdx.x;
  if (idx >= MROWS * DINNER) return;
  const float v = dt[idx] + bdt[idx & (DINNER - 1)];
  dt[idx] = (v > 20.f) ? v : log1pf(__expf(v));
}

// ---------------------------------------------------------------------------
// Selective scan + D-skip + SiLU(z) gating, writes bf16 y.
// One thread per (batch, channel): h[16] in VGPRs, per-step B/C in LDS.
// ---------------------------------------------------------------------------
__global__ __launch_bounds__(256)
void scan_kernel(const float* __restrict__ dt, const float* __restrict__ xc,
                 const float* __restrict__ xdbl, const float* __restrict__ xz,
                 const float* __restrict__ A_log, const float* __restrict__ Dv,
                 __bf16* __restrict__ ybf, int dir) {
  const int b  = blockIdx.x >> 3;
  const int ch = ((blockIdx.x & 7) << 8) + threadIdx.x;
  float a[DSTATE], h[DSTATE];
#pragma unroll
  for (int n = 0; n < DSTATE; ++n) {
    a[n] = -__expf(A_log[ch * DSTATE + n]);
    h[n] = 0.f;
  }
  const float Dc = Dv[ch];
  __shared__ float sBC[2 * DSTATE];

  for (int step = 0; step < SEQLEN; ++step) {
    const int t = dir ? (SEQLEN - 1 - step) : step;
    const size_t r = (size_t)b * SEQLEN + t;
    if (threadIdx.x < 2 * DSTATE)
      sBC[threadIdx.x] = xdbl[r * XDBLN + DTRANK + threadIdx.x];
    __syncthreads();
    const float dtv = dt[r * DINNER + ch];
    const float u   = xc[r * DINNER + ch];
    const float z   = xz[r * (2 * DINNER) + DINNER + ch];
    const float du  = dtv * u;
    float y = 0.f;
#pragma unroll
    for (int n = 0; n < DSTATE; ++n) {
      h[n] = h[n] * __expf(dtv * a[n]) + du * sBC[n];
      y += h[n] * sBC[DSTATE + n];
    }
    y = (y + u * Dc) * (z / (1.f + __expf(-z)));
    ybf[r * DINNER + ch] = (__bf16)y;
    __syncthreads();
  }
}

// ---------------------------------------------------------------------------
// Host-side launch
// ---------------------------------------------------------------------------
static inline void launch_gemm(const __bf16* A, const __bf16* B, float* C,
                               int M, int N, int K, int lda, int ldb, int ldc,
                               float alpha, int accum, hipStream_t s) {
  dim3 g((N + 127) / 128, (M + 127) / 128);
  wmma_gemm_kernel<<<g, 256, 0, s>>>(A, B, C, M, N, K, lda, ldb, ldc, alpha, accum);
}

extern "C" void kernel_launch(void* const* d_in, const int* in_sizes, int n_in,
                              void* d_out, int out_size, void* d_ws, size_t ws_size,
                              hipStream_t stream) {
  const float* x    = (const float*)d_in[0];
  const float* ln_w = (const float*)d_in[1];
  const float* ln_b = (const float*)d_in[2];

  size_t off = 0;
  char* base = (char*)d_ws;
  auto carve = [&](size_t bytes) -> void* {
    void* p = base + off;
    off += (bytes + 255) & ~(size_t)255;
    return p;
  };
  __bf16* xnbf   = (__bf16*)carve((size_t)MROWS * DMODEL * 2);
  __bf16* Wbf    = (__bf16*)carve((size_t)(2 * DINNER) * DMODEL * 2);
  float*  xzbuf  = (float*) carve((size_t)MROWS * 2 * DINNER * 4);
  float*  xcbuf  = (float*) carve((size_t)MROWS * DINNER * 4);
  __bf16* xcbf   = (__bf16*)carve((size_t)MROWS * DINNER * 2);
  float*  xdbl   = (float*) carve((size_t)MROWS * XDBLN * 4);
  __bf16* xdblbf = (__bf16*)carve((size_t)MROWS * XDBLN * 2);
  float*  dtbuf  = (float*) carve((size_t)MROWS * DINNER * 4);
  __bf16* ybf    = (__bf16*)carve((size_t)MROWS * DINNER * 2);

  // out = x (residual); both directions accumulate 0.5*y@W_out^T on top.
  (void)hipMemcpyAsync(d_out, x, (size_t)MROWS * DMODEL * sizeof(float),
                       hipMemcpyDeviceToDevice, stream);

  ln_bf16_kernel<<<MROWS, 256, 0, stream>>>(x, ln_w, ln_b, xnbf);

  const int cvt_blk = 256;
  for (int dir = 0; dir < 2; ++dir) {
    const int p = 3 + dir * 9;
    const float* W_in   = (const float*)d_in[p + 0];
    const float* W_conv = (const float*)d_in[p + 1];
    const float* b_conv = (const float*)d_in[p + 2];
    const float* W_x    = (const float*)d_in[p + 3];
    const float* W_dt   = (const float*)d_in[p + 4];
    const float* b_dt   = (const float*)d_in[p + 5];
    const float* A_log  = (const float*)d_in[p + 6];
    const float* Dvec   = (const float*)d_in[p + 7];
    const float* W_out  = (const float*)d_in[p + 8];

    // 1) xz = xn @ W_in^T : [8192, 4096]
    {
      const int n = 2 * DINNER * DMODEL;
      cvt_bf16_kernel<<<(n + cvt_blk - 1) / cvt_blk, cvt_blk, 0, stream>>>(W_in, Wbf, n);
      launch_gemm(xnbf, Wbf, xzbuf, MROWS, 2 * DINNER, DMODEL,
                  DMODEL, DMODEL, 2 * DINNER, 1.f, 0, stream);
    }

    // 2) xc = silu(depthwise_conv(xz[:, :DINNER]) + b_conv)
    {
      const int n = MROWS * DINNER;
      conv_silu_kernel<<<(n + 255) / 256, 256, 0, stream>>>(
          xzbuf, W_conv, b_conv, xcbuf, xcbf, dir);
    }

    // 3) x_dbl = xc @ W_x^T : [8192, 96]
    {
      const int n = XDBLN * DINNER;
      cvt_bf16_kernel<<<(n + cvt_blk - 1) / cvt_blk, cvt_blk, 0, stream>>>(W_x, Wbf, n);
      launch_gemm(xcbf, Wbf, xdbl, MROWS, XDBLN, DINNER,
                  DINNER, DINNER, XDBLN, 1.f, 0, stream);
    }

    // 4) dt_pre = x_dbl[:, :64] @ W_dt^T : [8192, 2048]
    {
      int n = MROWS * XDBLN;
      cvt_bf16_kernel<<<(n + cvt_blk - 1) / cvt_blk, cvt_blk, 0, stream>>>(xdbl, xdblbf, n);
      n = DINNER * DTRANK;
      cvt_bf16_kernel<<<(n + cvt_blk - 1) / cvt_blk, cvt_blk, 0, stream>>>(W_dt, Wbf, n);
      launch_gemm(xdblbf, Wbf, dtbuf, MROWS, DINNER, DTRANK,
                  XDBLN, DTRANK, DINNER, 1.f, 0, stream);
    }

    // 5) dt = softplus(dt_pre + b_dt)
    {
      const int n = MROWS * DINNER;
      softplus_kernel<<<(n + 255) / 256, 256, 0, stream>>>(dtbuf, b_dt);
    }

    // 6) selective scan + D-skip + SiLU(z) gating -> ybf (bf16)
    scan_kernel<<<32, 256, 0, stream>>>(dtbuf, xcbuf, xdbl, xzbuf,
                                        A_log, Dvec, ybf, dir);

    // 7) out += 0.5 * (y @ W_out^T)
    {
      const int n = DMODEL * DINNER;
      cvt_bf16_kernel<<<(n + cvt_blk - 1) / cvt_blk, cvt_blk, 0, stream>>>(W_out, Wbf, n);
      launch_gemm(ybf, Wbf, (float*)d_out, MROWS, DMODEL, DINNER,
                  DINNER, DINNER, DMODEL, 0.5f, 1, stream);
    }
  }
  (void)in_sizes; (void)n_in; (void)out_size; (void)ws_size;
}
